// TractableDendriticCell_16956530885004
// MI455X (gfx1250) — matrix operand order repaired and checked by reading, not compiled, not run on hardware
//
#include <hip/hip_runtime.h>
#include <cstdint>
#include <cstddef>

// Problem constants (from the reference)
#define B_    8192
#define IN_   1024
#define HID_  2560
#define CS_   512
#define K1_   (IN_ + HID_)   // 3584

typedef __attribute__((ext_vector_type(16))) __bf16 v16bf;
typedef __attribute__((ext_vector_type(8)))  __bf16 v8bf;
typedef __attribute__((ext_vector_type(8)))  float  v8f;

// ---- fp32 <-> bf16 split helpers (round-to-nearest-even, finite inputs) ----
__device__ __forceinline__ unsigned short f2bf(float f) {
  unsigned u = __float_as_uint(f);
  u += 0x7FFFu + ((u >> 16) & 1u);
  return (unsigned short)(u >> 16);
}
__device__ __forceinline__ float bf2f(unsigned short h) {
  return __uint_as_float(((unsigned)h) << 16);
}

// -------------------------------------------------------------------------
// Pack W_cat^T = [W_in | W_rec] transposed to [N=2560][K=3584] bf16 hi/lo.
// Row n (= c*512 + o) holds K contiguous -> B tiles stream as 16B chunks.
// -------------------------------------------------------------------------
__global__ __launch_bounds__(256) void pack_w1(const float* __restrict__ W_in,
                                               const float* __restrict__ W_rec,
                                               unsigned short* __restrict__ hi,
                                               unsigned short* __restrict__ lo) {
  size_t idx = (size_t)blockIdx.x * 256 + threadIdx.x;
  if (idx >= (size_t)HID_ * K1_) return;
  int n = (int)(idx / K1_);
  int k = (int)(idx % K1_);
  int c = n >> 9;          // n / 512
  int o = n & 511;         // n % 512
  float w = (k < IN_) ? W_in[((size_t)c * IN_ + k) * CS_ + o]
                      : W_rec[((size_t)c * HID_ + (k - IN_)) * CS_ + o];
  unsigned short hv = f2bf(w);
  hi[idx] = hv;
  lo[idx] = f2bf(w - bf2f(hv));
}

// Pack W_int^T to [N=2560][K=2560] bf16 hi/lo. Strided reads hit L2 (26MB << 192MB).
__global__ __launch_bounds__(256) void pack_w2(const float* __restrict__ W_int,
                                               unsigned short* __restrict__ hi,
                                               unsigned short* __restrict__ lo) {
  size_t idx = (size_t)blockIdx.x * 256 + threadIdx.x;
  if (idx >= (size_t)HID_ * HID_) return;
  int n = (int)(idx / HID_);
  int k = (int)(idx % HID_);
  float w = W_int[(size_t)k * HID_ + n];
  unsigned short hv = f2bf(w);
  hi[idx] = hv;
  lo[idx] = f2bf(w - bf2f(hv));
}

// -------------------------------------------------------------------------
// C = tanh(A * B^T_packed + bias0 [+ bias1])
//   A: fp32 row-major, K split across two sources (A0 for k<kA0, else A1);
//      staged through VGPRs with on-the-fly fp32 -> bf16 hi/lo split.
//   B: pre-packed bf16 hi/lo [N][K]; streamed straight into LDS with
//      global_load_async_to_lds_b128 (ASYNCcnt) -- zero VGPR footprint.
// Split-bf16 ("bf16x3"): acc += Ah*Bh + Al*Bh + Ah*Bl, fp32 accumulate.
// Block tile 128x128x32, 8 wave32s, wave tile 64x32 (4x2 WMMA tiles).
// Double-buffered LDS, 80B row pitch (16B aligned, conflict-free).
// -------------------------------------------------------------------------
__global__ __launch_bounds__(256) void gemm_bf16x3_bias_tanh(
    const float* __restrict__ A0, int ldA0, int kA0,
    const float* __restrict__ A1, int ldA1,
    const unsigned short* __restrict__ Bth,
    const unsigned short* __restrict__ Btl,
    const float* __restrict__ bias0,
    const float* __restrict__ bias1,
    float* __restrict__ C,
    int N, int K) {
  __shared__ alignas(16) unsigned short sAh[2][128 * 40];
  __shared__ alignas(16) unsigned short sAl[2][128 * 40];
  __shared__ alignas(16) unsigned short sBh[2][128 * 40];
  __shared__ alignas(16) unsigned short sBl[2][128 * 40];

  const int tid   = threadIdx.x;
  const int lane  = tid & 31;
  const int wave  = tid >> 5;
  const int wm    = wave & 1;   // 0..1 : 64-row half of M tile
  const int wn    = wave >> 1;  // 0..3 : 32-col slice of N tile
  const int lrow  = lane & 15;
  const int lhalf = lane >> 4;
  const int mBase = blockIdx.y * 128;
  const int nBase = blockIdx.x * 128;
  const int T     = K >> 5;     // K / 32

  float4 ra[4];

  // B tile: async copy global -> LDS (hi & lo), 4 x b128 per thread.
  // vdst = per-lane LDS byte address (low 32 bits of the generic pointer),
  // vaddr = per-lane byte offset, saddr = uniform 64-bit base.
  auto issueB = [&](int kt, int buf) {
    const int kg = kt << 5;
#pragma unroll
    for (int i = 0; i < 2; ++i) {
      int idx = tid + (i << 8);
      int n = idx >> 2, kc = idx & 3;
      unsigned goff =
          (unsigned)(((size_t)(nBase + n) * K + kg + (kc << 3)) * sizeof(unsigned short));
      unsigned ldsH = (unsigned)(uintptr_t)&sBh[buf][n * 40 + (kc << 3)];
      unsigned ldsL = (unsigned)(uintptr_t)&sBl[buf][n * 40 + (kc << 3)];
      asm volatile("global_load_async_to_lds_b128 %0, %1, %2"
                   :: "v"(ldsH), "v"(goff), "s"(Bth) : "memory");
      asm volatile("global_load_async_to_lds_b128 %0, %1, %2"
                   :: "v"(ldsL), "v"(goff), "s"(Btl) : "memory");
    }
  };

  // A tile: global -> registers (fp32), 4 float4 per thread.
  auto fetchA = [&](int kt) {
    const int kg = kt << 5;
    const float* Ap;
    int ldA, k0;
    if (kg < kA0) { Ap = A0; ldA = ldA0; k0 = kg; }
    else          { Ap = A1; ldA = ldA1; k0 = kg - kA0; }
#pragma unroll
    for (int i = 0; i < 4; ++i) {
      int idx = tid + (i << 8);
      int row = idx >> 3, kq = idx & 7;
      ra[i] = *reinterpret_cast<const float4*>(
          Ap + (size_t)(mBase + row) * ldA + k0 + (kq << 2));
    }
  };

  // A registers -> LDS with fp32 -> bf16 hi/lo split.
  auto stageA = [&](int b) {
#pragma unroll
    for (int i = 0; i < 4; ++i) {
      int idx = tid + (i << 8);
      int row = idx >> 3, kq = idx & 7;
      float4 v = ra[i];
      unsigned short h0 = f2bf(v.x), h1 = f2bf(v.y), h2 = f2bf(v.z), h3 = f2bf(v.w);
      unsigned short l0 = f2bf(v.x - bf2f(h0)), l1 = f2bf(v.y - bf2f(h1));
      unsigned short l2 = f2bf(v.z - bf2f(h2)), l3 = f2bf(v.w - bf2f(h3));
      uint2 hp = make_uint2((unsigned)h0 | ((unsigned)h1 << 16),
                            (unsigned)h2 | ((unsigned)h3 << 16));
      uint2 lp = make_uint2((unsigned)l0 | ((unsigned)l1 << 16),
                            (unsigned)l2 | ((unsigned)l3 << 16));
      *reinterpret_cast<uint2*>(&sAh[b][row * 40 + (kq << 2)]) = hp;
      *reinterpret_cast<uint2*>(&sAl[b][row * 40 + (kq << 2)]) = lp;
    }
  };

  v8f acc[4][2] = {};

  issueB(0, 0);
  fetchA(0);
  stageA(0);
  asm volatile("s_wait_asynccnt 0" ::: "memory");
  __syncthreads();

  for (int kt = 0; kt < T; ++kt) {
    const int cur = kt & 1;
    if (kt + 1 < T) {
      issueB(kt + 1, cur ^ 1);  // other buffer: last read before previous barrier
      fetchA(kt + 1);
    }
    if (kt + 2 < T) {  // L2 prefetch of the K+2 A tile (B rides the async engine)
      const int kg2 = (kt + 2) << 5;
      const float* Ap;
      int ldA, k0;
      if (kg2 < kA0) { Ap = A0; ldA = ldA0; k0 = kg2; }
      else           { Ap = A1; ldA = ldA1; k0 = kg2 - kA0; }
      int row = tid >> 1;
      __builtin_prefetch(Ap + (size_t)(mBase + row) * ldA + k0 + ((tid & 1) << 4), 0, 1);
    }

    // B operands: lane holds column n = lane%16, K-run [16*lhalf, +16) contiguous
    v16bf bh[2], bl[2];
#pragma unroll
    for (int j = 0; j < 2; ++j) {
      int base = (wn * 32 + j * 16 + lrow) * 40 + (lhalf << 4);
      v8bf p0 = *reinterpret_cast<const v8bf*>(&sBh[cur][base]);
      v8bf p1 = *reinterpret_cast<const v8bf*>(&sBh[cur][base + 8]);
      bh[j] = __builtin_shufflevector(p0, p1, 0, 1, 2, 3, 4, 5, 6, 7,
                                      8, 9, 10, 11, 12, 13, 14, 15);
      v8bf q0 = *reinterpret_cast<const v8bf*>(&sBl[cur][base]);
      v8bf q1 = *reinterpret_cast<const v8bf*>(&sBl[cur][base + 8]);
      bl[j] = __builtin_shufflevector(q0, q1, 0, 1, 2, 3, 4, 5, 6, 7,
                                      8, 9, 10, 11, 12, 13, 14, 15);
    }
    // A operands loaded per m-tile (keeps only one ah/al pair live -> no spills).
    // Lane holds row m = lane%16; K-run [16*lhalf, +16) contiguous.
#pragma unroll
    for (int i = 0; i < 4; ++i) {
      int base = (wm * 64 + i * 16 + lrow) * 40;
      v8bf p0 = *reinterpret_cast<const v8bf*>(&sAh[cur][base + (lhalf << 4)]);
      v8bf p1 = *reinterpret_cast<const v8bf*>(&sAh[cur][base + (lhalf << 4) + 8]);
      v16bf ah = __builtin_shufflevector(p0, p1, 0, 1, 2, 3, 4, 5, 6, 7,
                                         8, 9, 10, 11, 12, 13, 14, 15);
      v8bf q0 = *reinterpret_cast<const v8bf*>(&sAl[cur][base + (lhalf << 4)]);
      v8bf q1 = *reinterpret_cast<const v8bf*>(&sAl[cur][base + (lhalf << 4) + 8]);
      v16bf al = __builtin_shufflevector(q0, q1, 0, 1, 2, 3, 4, 5, 6, 7,
                                         8, 9, 10, 11, 12, 13, 14, 15);
      // bf16x3 split-precision accumulation: Ah*Bh + Al*Bh + Ah*Bl
#pragma unroll
      for (int j = 0; j < 2; ++j) {
        acc[i][j] = __builtin_amdgcn_wmma_f32_16x16x32_bf16(
            false, ah, false, bh[j], (short)0, acc[i][j], false, false);
        acc[i][j] = __builtin_amdgcn_wmma_f32_16x16x32_bf16(
            false, al, false, bh[j], (short)0, acc[i][j], false, false);
        acc[i][j] = __builtin_amdgcn_wmma_f32_16x16x32_bf16(
            false, ah, false, bl[j], (short)0, acc[i][j], false, false);
      }
    }

    if (kt + 1 < T) {
      stageA(cur ^ 1);
      asm volatile("s_wait_asynccnt 0" ::: "memory");
      __syncthreads();
    }
  }

  // Epilogue: bias + tanh, C/D layout m = 8*lhalf + v, n = lane%16
#pragma unroll
  for (int j = 0; j < 2; ++j) {
    int ng = nBase + wn * 32 + j * 16 + lrow;
    float bsum = bias0[ng] + (bias1 ? bias1[ng] : 0.0f);
#pragma unroll
    for (int i = 0; i < 4; ++i) {
      int mg = mBase + wm * 64 + i * 16 + (lhalf << 3);
#pragma unroll
      for (int v = 0; v < 8; ++v) {
        C[(size_t)(mg + v) * N + ng] = tanhf(acc[i][j][v] + bsum);
      }
    }
  }
}

// -------------------------------------------------------------------------
// Launcher: pack weights (bf16 hi/lo, transposed) -> GEMM1 -> GEMM2.
// Workspace layout (needs ~147 MB):
//   Wt1 hi/lo : 2 x 2560*3584*2B = 36.7 MB
//   Wt2 hi/lo : 2 x 2560*2560*2B = 26.2 MB
//   C1        : 8192*2560*4B     = 83.9 MB
// -------------------------------------------------------------------------
extern "C" void kernel_launch(void* const* d_in, const int* in_sizes, int n_in,
                              void* d_out, int out_size, void* d_ws, size_t ws_size,
                              hipStream_t stream) {
  (void)in_sizes; (void)n_in; (void)out_size; (void)ws_size;
  const float* x     = (const float*)d_in[0];
  const float* h     = (const float*)d_in[1];
  const float* W_in  = (const float*)d_in[2];
  const float* b_in  = (const float*)d_in[3];
  const float* W_rec = (const float*)d_in[4];
  const float* b_rec = (const float*)d_in[5];
  const float* W_int = (const float*)d_in[6];
  const float* b_int = (const float*)d_in[7];
  float* out = (float*)d_out;
  char*  ws  = (char*)d_ws;

  const size_t szWt1 = (size_t)HID_ * K1_ * sizeof(unsigned short);
  const size_t szWt2 = (size_t)HID_ * HID_ * sizeof(unsigned short);
  unsigned short* Wt1h = (unsigned short*)(ws);
  unsigned short* Wt1l = (unsigned short*)(ws + szWt1);
  unsigned short* Wt2h = (unsigned short*)(ws + 2 * szWt1);
  unsigned short* Wt2l = (unsigned short*)(ws + 2 * szWt1 + szWt2);
  float*          C1   = (float*)(ws + 2 * szWt1 + 2 * szWt2);

  pack_w1<<<((unsigned)((size_t)HID_ * K1_ + 255) / 256), 256, 0, stream>>>(
      W_in, W_rec, Wt1h, Wt1l);
  pack_w2<<<((unsigned)((size_t)HID_ * HID_ + 255) / 256), 256, 0, stream>>>(
      W_int, Wt2h, Wt2l);

  // GEMM1: [8192 x 3584] * [3584 x 2560], A = [x | h] (K split at 1024)
  gemm_bf16x3_bias_tanh<<<dim3(HID_ / 128, B_ / 128), 256, 0, stream>>>(
      x, IN_, IN_, h, HID_, Wt1h, Wt1l, b_in, b_rec, C1, HID_, K1_);

  // GEMM2: [8192 x 2560] * [2560 x 2560]
  gemm_bf16x3_bias_tanh<<<dim3(HID_ / 128, B_ / 128), 256, 0, stream>>>(
      C1, HID_, HID_, nullptr, 0, Wt2h, Wt2l, b_int, nullptr, out, HID_, HID_);
}
